// DeltaNet_31877247271440
// MI455X (gfx1250) — matrix operand order repaired and checked by reading, not compiled, not run on hardware
//
#include <hip/hip_runtime.h>

// ---------------- problem constants ----------------
#define BB 2
#define LL 4096
#define DD 1024
#define HH 4
#define DKC 256
#define DVC 256
#define CHK 32
#define NCH (LL / CHK)      // 128 chunks
#define DVS 64              // dv slice handled per delta block
#define NSL (DVC / DVS)     // 4 slices

typedef __attribute__((ext_vector_type(16))) _Float16 v16h;
typedef __attribute__((ext_vector_type(8)))  float    v8f;
typedef __attribute__((ext_vector_type(4)))  int      i32x4;

union Frag { unsigned u[8]; v16h h; };
union H2   { _Float16 h[2]; unsigned u; };
union H8   { uint4 v; _Float16 h[8]; };

__device__ inline v8f wmma_f16(v16h a, v16h b, v8f c) {
  // D = A(16x32 f16) * B(32x16 f16) + C(f32), v_wmma_f32_16x16x32_f16
  return __builtin_amdgcn_wmma_f32_16x16x32_f16(false, a, false, b, (short)0, c, false, false);
}
__device__ inline unsigned pack2(_Float16 a, _Float16 b) { H2 t; t.h[0] = a; t.h[1] = b; return t.u; }

// ---------------- CDNA5 async global->LDS copy (ASYNCcnt-tracked DMA) ----------------
typedef __attribute__((address_space(1))) i32x4* gp1_t;
typedef __attribute__((address_space(3))) i32x4* lp3_t;

__device__ inline void async_copy16(const void* g, void* l) {
#if __has_builtin(__builtin_amdgcn_global_load_async_to_lds_b128)
  __builtin_amdgcn_global_load_async_to_lds_b128((gp1_t)(uintptr_t)g, (lp3_t)(uintptr_t)l, 0, 0);
#else
  // LDS byte offset = low 32 bits of the generic LDS address (flat->LDS truncation)
  unsigned loff = (unsigned)(uintptr_t)l;
  unsigned long long ga = (unsigned long long)(uintptr_t)g;
  asm volatile("global_load_async_to_lds_b128 %0, %1, off" :: "v"(loff), "v"(ga) : "memory");
#endif
}
__device__ inline void wait_async0() {
#if __has_builtin(__builtin_amdgcn_s_wait_asynccnt)
  __builtin_amdgcn_s_wait_asynccnt(0);
#else
  asm volatile("s_wait_asynccnt 0" ::: "memory");
#endif
}

// A-operand frag from row-major LDS f16 [.. x ld], tile at (m0, k0), K=32.
// ISA layout: lane = M (lane&15); lanes0-15 K={0..7,16..23}, lanes16-31 K={8..15,24..31}
__device__ inline v16h frag_A(const _Float16* m, int ld, int m0, int k0, int lane) {
  Frag f;
  const unsigned* row = (const unsigned*)(m + (size_t)(m0 + (lane & 15)) * ld + k0);
  const int half4 = (lane >> 4) << 2;
#pragma unroll
  for (int v = 0; v < 8; v++) f.u[v] = row[((v >> 2) << 3) + half4 + (v & 3)];
  return f.h;
}
// B-operand frag from row-major LDS f16 [K x ld] at (k0, n0): lane = N (lane&15); lanes0-15 K={0..7,16..23}, lanes16-31 K={8..15,24..31}
__device__ inline v16h frag_B(const _Float16* m, int ld, int k0, int n0, int lane) {
  Frag f;
  const int nn = n0 + (lane & 15);
  const int half8 = (lane >> 4) * 8;
#pragma unroll
  for (int v = 0; v < 8; v++) {
    int K = k0 + ((v >> 2) << 4) + half8 + ((v & 3) << 1);
    f.u[v] = pack2(m[(size_t)K * ld + nn], m[(size_t)(K + 1) * ld + nn]);
  }
  return f.h;
}
// B = M^T where M is row-major [N x ld]: B[k][n] = M[n][k]
__device__ inline v16h frag_BT(const _Float16* m, int ld, int k0, int n0, int lane) {
  Frag f;
  const unsigned* row = (const unsigned*)(m + (size_t)(n0 + (lane & 15)) * ld + k0);
  const int half4 = (lane >> 4) << 2;
#pragma unroll
  for (int v = 0; v < 8; v++) f.u[v] = row[((v >> 2) << 3) + half4 + (v & 3)];
  return f.h;
}
// A = M^T where M is row-major [K x ld]: A[mm][k] = M[k][mm]
__device__ inline v16h frag_AT(const _Float16* m, int ld, int m0, int k0, int lane) {
  Frag f;
  const int mm = m0 + (lane & 15);
  const int half8 = (lane >> 4) * 8;
#pragma unroll
  for (int v = 0; v < 8; v++) {
    int K = k0 + ((v >> 2) << 4) + half8 + ((v & 3) << 1);
    f.u[v] = pack2(m[(size_t)K * ld + mm], m[(size_t)(K + 1) * ld + mm]);
  }
  return f.h;
}

// ---------------- cast f32 -> f16 ----------------
__global__ void cast_f16(const float* __restrict__ in, _Float16* __restrict__ out, size_t n) {
  size_t i = (size_t)blockIdx.x * blockDim.x + threadIdx.x;
  if (i < n) out[i] = (_Float16)in[i];
}

// ---------------- WMMA GEMM: C[M,N] f32 = A[M,K] f16 * B[K,N] f16 ----------------
// block = 256 threads (8 waves), tile 128x128, K-step 32; wave computes 32x64.
// Double-buffered LDS tiles filled by async global->LDS DMA: tile t+1 streams in
// while tile t feeds the WMMAs; one s_wait_asynccnt + barrier per K-step.
__global__ __launch_bounds__(256) void gemm_f16(const _Float16* __restrict__ A,
                                                const _Float16* __restrict__ Bm,
                                                float* __restrict__ C, int M, int N, int K) {
  __shared__ _Float16 sA[2][128 * 32];
  __shared__ _Float16 sB[2][32 * 128];
  const int m0 = blockIdx.y * 128, n0 = blockIdx.x * 128;
  const int tid = threadIdx.x, lane = tid & 31, wave = tid >> 5;
  const int wm = wave & 3, wn = wave >> 2;
  const v8f z = {0.f, 0.f, 0.f, 0.f, 0.f, 0.f, 0.f, 0.f};
  v8f acc[2][4];
#pragma unroll
  for (int i = 0; i < 2; i++)
#pragma unroll
    for (int j = 0; j < 4; j++) acc[i][j] = z;

  const int ra = tid >> 1, sa = (tid & 1) * 16;   // A stage coords (128x32)
  const int rb = tid >> 3, sb = (tid & 7) * 16;   // B stage coords (32x128)

  // issue tile 0
  async_copy16(A + (size_t)(m0 + ra) * K + 0 + sa,      &sA[0][ra * 32 + sa]);
  async_copy16(A + (size_t)(m0 + ra) * K + 0 + sa + 8,  &sA[0][ra * 32 + sa + 8]);
  async_copy16(Bm + (size_t)(0 + rb) * N + n0 + sb,     &sB[0][rb * 128 + sb]);
  async_copy16(Bm + (size_t)(0 + rb) * N + n0 + sb + 8, &sB[0][rb * 128 + sb + 8]);

  int buf = 0;
  for (int kt = 0; kt < K; kt += 32, buf ^= 1) {
    wait_async0();
    __syncthreads();
    if (kt + 32 < K) { // stream next tile into the other buffer while computing
      int nb = buf ^ 1, kn = kt + 32;
      async_copy16(A + (size_t)(m0 + ra) * K + kn + sa,      &sA[nb][ra * 32 + sa]);
      async_copy16(A + (size_t)(m0 + ra) * K + kn + sa + 8,  &sA[nb][ra * 32 + sa + 8]);
      async_copy16(Bm + (size_t)(kn + rb) * N + n0 + sb,     &sB[nb][rb * 128 + sb]);
      async_copy16(Bm + (size_t)(kn + rb) * N + n0 + sb + 8, &sB[nb][rb * 128 + sb + 8]);
    }
    v16h fa[2], fb[4];
#pragma unroll
    for (int mt = 0; mt < 2; mt++) fa[mt] = frag_A(sA[buf], 32, wm * 32 + mt * 16, 0, lane);
#pragma unroll
    for (int nt = 0; nt < 4; nt++) fb[nt] = frag_B(sB[buf], 128, 0, wn * 64 + nt * 16, lane);
#pragma unroll
    for (int mt = 0; mt < 2; mt++)
#pragma unroll
      for (int nt = 0; nt < 4; nt++) acc[mt][nt] = wmma_f16(fa[mt], fb[nt], acc[mt][nt]);
  }
  const int half8 = (lane >> 4) * 8, nl = lane & 15;
#pragma unroll
  for (int mt = 0; mt < 2; mt++)
#pragma unroll
    for (int nt = 0; nt < 4; nt++)
#pragma unroll
      for (int r = 0; r < 8; r++) {
        int row = m0 + wm * 32 + mt * 16 + r + half8;
        int col = n0 + wn * 64 + nt * 16 + nl;
        C[(size_t)row * N + col] = acc[mt][nt][r];
      }
}

// ---------------- causal depthwise conv (K=4) + SiLU, f32 in -> f16 out ----------------
__global__ void conv_silu(const float* __restrict__ xl, const float* __restrict__ w,
                          _Float16* __restrict__ out) {
  size_t idx = (size_t)blockIdx.x * blockDim.x + threadIdx.x;
  if (idx >= (size_t)BB * LL * DD) return;
  int c = (int)(idx % DD);
  size_t bl = idx / DD;
  int l = (int)(bl % LL);
  float acc = 0.f;
#pragma unroll
  for (int t = 0; t < 4; t++) {
    int ls = l - 3 + t;
    if (ls >= 0) acc += xl[(bl - (size_t)l + (size_t)ls) * DD + c] * w[c * 4 + t];
  }
  out[idx] = (_Float16)(acc / (1.f + expf(-acc)));
}

// ---------------- small projections: beta=sig(xWb), wg=sig(xWg+bg), logits=xWl+bl ----------------
__global__ void small_proj(const float* __restrict__ x, const float* __restrict__ Wb,
                           const float* __restrict__ Wg, const float* __restrict__ bg,
                           const float* __restrict__ Wl, const float* __restrict__ bl,
                           float* __restrict__ beta, float* __restrict__ wg,
                           float* __restrict__ lg) {
  int idx = blockIdx.x * blockDim.x + threadIdx.x;
  if (idx >= BB * LL * 20) return;
  int j = idx % 20;
  size_t r = (size_t)(idx / 20);
  const float* xr = x + r * DD;
  float acc = 0.f;
  if (j < 4) {
    for (int d = 0; d < DD; d++) acc += xr[d] * Wb[d * 4 + j];
    beta[r * 4 + j] = 1.f / (1.f + expf(-acc));
  } else if (j < 8) {
    int g = j - 4;
    for (int d = 0; d < DD; d++) acc += xr[d] * Wg[d * 4 + g];
    wg[r * 4 + g] = 1.f / (1.f + expf(-(acc + bg[g])));
  } else {
    int q = j - 8;
    for (int d = 0; d < DD; d++) acc += xr[d] * Wl[d * 12 + q];
    lg[r * 12 + q] = acc + bl[q];
  }
}

// ---------------- delta rule, chunkwise, WMMA everywhere ----------------
// grid (NSL, H, B), block 256 (8 waves). State S[256 dk x 64 dv]: f32 master in wave
// accumulators (wave owns dk strips {w,w+8}), f16 mirror in LDS as WMMA B-operand.
// 320KB/WGP LDS keeps the whole recurrent state + chunk tiles on-chip.
__global__ __launch_bounds__(256) void delta_kernel(const _Float16* __restrict__ qa,
                                                    const _Float16* __restrict__ ka,
                                                    const _Float16* __restrict__ va,
                                                    const float* __restrict__ beta,
                                                    float* __restrict__ dout) {
  extern __shared__ char smem[];
  _Float16* qc  = (_Float16*)smem;        // 32x256
  _Float16* kc  = qc  + 32 * 256;         // 32x256
  _Float16* kbc = kc  + 32 * 256;         // 32x256
  _Float16* wld = kbc + 32 * 256;         // w = T@kbc, 32x256
  _Float16* vb  = wld + 32 * 256;         // 32x64 (v*beta, dv slice)
  _Float16* ul  = vb  + 32 * 64;          // u, 32x64
  _Float16* at  = ul  + 32 * 64;          // attn (masked), 32x32
  _Float16* Th  = at  + 32 * 32;          // T f16, 32x32
  _Float16* Sh  = Th  + 32 * 32;          // S f16 mirror, 256x64
  float*    Af  = (float*)(Sh + 256 * 64);// A f32, 32x32
  float*    Tf  = Af + 32 * 32;           // T f32, 32x32

  const int dv0 = blockIdx.x * DVS;
  const int h = blockIdx.y, b = blockIdx.z;
  const int tid = threadIdx.x, lane = tid & 31, wave = tid >> 5;
  const int half8 = (lane >> 4) * 8, nl = lane & 15;
  const int mt = wave >> 2, nt = wave & 3;  // per-wave (m-tile, n-tile) task for 32x64 outputs
  const v8f z = {0.f, 0.f, 0.f, 0.f, 0.f, 0.f, 0.f, 0.f};

  { // zero S mirror
    unsigned* p = (unsigned*)Sh;
    for (int i = tid; i < 256 * 64 / 2; i += 256) p[i] = 0u;
  }
  v8f Sacc[2][4];
#pragma unroll
  for (int i = 0; i < 2; i++)
#pragma unroll
    for (int j = 0; j < 4; j++) Sacc[i][j] = z;
  __syncthreads();

  for (int ch = 0; ch < NCH; ch++) {
    const int l0 = ch * CHK;
    // ---- P1: load chunk, l2norm q/k, apply beta ----
#pragma unroll
    for (int rr = 0; rr < 4; rr++) {
      int r = wave * 4 + rr;
      size_t base = (size_t)(b * LL + l0 + r) * DD + h * DKC;
      H8 qv, kv;
      qv.v = *(const uint4*)(qa + base + lane * 8);
      kv.v = *(const uint4*)(ka + base + lane * 8);
      float qf[8], kf[8], sq = 0.f, sk = 0.f;
#pragma unroll
      for (int i = 0; i < 8; i++) {
        qf[i] = (float)qv.h[i]; sq += qf[i] * qf[i];
        kf[i] = (float)kv.h[i]; sk += kf[i] * kf[i];
      }
#pragma unroll
      for (int off = 16; off > 0; off >>= 1) {
        sq += __shfl_xor(sq, off, 32);
        sk += __shfl_xor(sk, off, 32);
      }
      float qs = rsqrtf(sq + 1e-6f), ks = rsqrtf(sk + 1e-6f);
      float bet = beta[(size_t)(b * LL + l0 + r) * HH + h];
      H8 oq, ok, ob;
#pragma unroll
      for (int i = 0; i < 8; i++) {
        oq.h[i] = (_Float16)(qf[i] * qs);
        ok.h[i] = (_Float16)(kf[i] * ks);
        ob.h[i] = (_Float16)(kf[i] * ks * bet);
      }
      *(uint4*)(qc  + r * 256 + lane * 8) = oq.v;
      *(uint4*)(kc  + r * 256 + lane * 8) = ok.v;
      *(uint4*)(kbc + r * 256 + lane * 8) = ob.v;
      H2 vv; vv.u = *(const unsigned*)(va + base + dv0 + lane * 2);
      H2 ov;
      ov.h[0] = (_Float16)((float)vv.h[0] * bet);
      ov.h[1] = (_Float16)((float)vv.h[1] * bet);
      *(unsigned*)(vb + r * 64 + lane * 2) = ov.u;
    }
    __syncthreads();
    // ---- P2: A = kbc @ kc^T (waves 0-3), attn = tril(qc @ kc^T) (waves 4-7) ----
    {
      int w4 = wave & 3, tmt = w4 >> 1, tnt = w4 & 1;
      v8f a = z;
      if (wave < 4) {
#pragma unroll
        for (int kk = 0; kk < 8; kk++)
          a = wmma_f16(frag_A(kbc, 256, tmt * 16, kk * 32, lane),
                       frag_BT(kc, 256, kk * 32, tnt * 16, lane), a);
#pragma unroll
        for (int r = 0; r < 8; r++) Af[(tmt * 16 + r + half8) * 32 + tnt * 16 + nl] = a[r];
      } else {
#pragma unroll
        for (int kk = 0; kk < 8; kk++)
          a = wmma_f16(frag_A(qc, 256, tmt * 16, kk * 32, lane),
                       frag_BT(kc, 256, kk * 32, tnt * 16, lane), a);
#pragma unroll
        for (int r = 0; r < 8; r++) {
          int ro = tmt * 16 + r + half8, co = tnt * 16 + nl;
          at[ro * 32 + co] = (co <= ro) ? (_Float16)a[r] : (_Float16)0.f;
        }
      }
    }
    __syncthreads();
    // ---- P3: T = (I + tril(A,-1))^-1 via forward substitution (wave 0, lane = column) ----
    if (wave == 0) {
      int j = lane;
      for (int i = 0; i < 32; i++) {
        float t = (i == j) ? 1.f : 0.f;
        for (int m2 = 0; m2 < i; m2++) t -= Af[i * 32 + m2] * Tf[m2 * 32 + j];
        Tf[i * 32 + j] = t;
        Th[i * 32 + j] = (_Float16)t;
      }
    }
    __syncthreads();
    // ---- P4: w = T @ kbc (full dk, 4 tiles/wave); u0 = T @ vb (per-wave tile, regs) ----
#pragma unroll
    for (int jj = 0; jj < 4; jj++) {
      int idx = wave * 4 + jj, wmt = idx >> 4, wct = idx & 15;
      v8f a = wmma_f16(frag_A(Th, 32, wmt * 16, 0, lane),
                       frag_B(kbc, 256, 0, wct * 16, lane), z);
#pragma unroll
      for (int r = 0; r < 8; r++) wld[(wmt * 16 + r + half8) * 256 + wct * 16 + nl] = (_Float16)a[r];
    }
    v8f accu = wmma_f16(frag_A(Th, 32, mt * 16, 0, lane), frag_B(vb, 64, 0, nt * 16, lane), z);
    __syncthreads();
    // ---- P5: u = u0 - w @ S ----
    {
      v8f a = z;
#pragma unroll
      for (int kk = 0; kk < 8; kk++)
        a = wmma_f16(frag_A(wld, 256, mt * 16, kk * 32, lane),
                     frag_B(Sh, 64, kk * 32, nt * 16, lane), a);
#pragma unroll
      for (int r = 0; r < 8; r++) accu[r] -= a[r];
#pragma unroll
      for (int r = 0; r < 8; r++) ul[(mt * 16 + r + half8) * 64 + nt * 16 + nl] = (_Float16)accu[r];
    }
    __syncthreads();
    // ---- P6: o = q @ S + attn @ u (write out); S += k^T @ u (register accums) ----
    {
      v8f a = z;
#pragma unroll
      for (int kk = 0; kk < 8; kk++)
        a = wmma_f16(frag_A(qc, 256, mt * 16, kk * 32, lane),
                     frag_B(Sh, 64, kk * 32, nt * 16, lane), a);
      a = wmma_f16(frag_A(at, 32, mt * 16, 0, lane), frag_B(ul, 64, 0, nt * 16, lane), a);
#pragma unroll
      for (int r = 0; r < 8; r++) {
        int row = mt * 16 + r + half8;
        int l = l0 + row, col = dv0 + nt * 16 + nl;
        dout[((size_t)(b * LL + l) * HH + h) * DVC + col] = a[r];
      }
    }
#pragma unroll
    for (int si = 0; si < 2; si++) {
      int s2 = wave + si * 8;
      v16h afr = frag_AT(kc, 256, s2 * 16, 0, lane);
#pragma unroll
      for (int n2 = 0; n2 < 4; n2++)
        Sacc[si][n2] = wmma_f16(afr, frag_B(ul, 64, 0, n2 * 16, lane), Sacc[si][n2]);
    }
    __syncthreads();
    // ---- P7: refresh S f16 mirror from accumulators ----
#pragma unroll
    for (int si = 0; si < 2; si++)
#pragma unroll
      for (int n2 = 0; n2 < 4; n2++)
#pragma unroll
        for (int r = 0; r < 8; r++)
          Sh[(size_t)((wave + si * 8) * 16 + r + half8) * 64 + n2 * 16 + nl] =
              (_Float16)Sacc[si][n2][r];
    __syncthreads();
  }
}

// ---------------- FIR paths + hierarchical gating + per-head RMSNorm ----------------
// grid (L/64, H, B), block 256 (thread = dv). v tile [94 x 256] staged in LDS.
__global__ __launch_bounds__(256) void fir_mix(const _Float16* __restrict__ va,
                                               const float* __restrict__ dout,
                                               const float* __restrict__ wg,
                                               const float* __restrict__ lgt,
                                               const float* __restrict__ fs,
                                               const float* __restrict__ fl,
                                               const float* __restrict__ log_temp,
                                               const float* __restrict__ rmsw,
                                               _Float16* __restrict__ oh) {
  extern __shared__ float sm[];
  float* vt = sm;              // 94*256
  float* red = sm + 94 * 256;  // 8
  const int l0 = blockIdx.x * 64, h = blockIdx.y, b = blockIdx.z;
  const int dv = threadIdx.x;
  for (int i = 0; i < 94; i++) {
    int l = l0 - 30 + i;
    vt[i * 256 + dv] = (l >= 0) ? (float)va[(size_t)(b * LL + l) * DD + h * DVC + dv] : 0.f;
  }
  float ws3[3], wl31[31];
#pragma unroll
  for (int t = 0; t < 3; t++) ws3[t] = fs[(h * DVC + dv) * 3 + t];
#pragma unroll
  for (int t = 0; t < 31; t++) wl31[t] = fl[(h * DVC + dv) * 31 + t];
  const float rw = rmsw[dv];
  const float itemp = 1.f / expf(log_temp[h]);
  __syncthreads();
  for (int i = 0; i < 64; i++) {
    const int l = l0 + i;
    float vcur = vt[(30 + i) * 256 + dv];
    float lsv = 0.f, llv = 0.f;
#pragma unroll
    for (int t = 0; t < 3; t++) lsv += vt[(28 + i + t) * 256 + dv] * ws3[t];
#pragma unroll
    for (int t = 0; t < 31; t++) llv += vt[(i + t) * 256 + dv] * wl31[t];
    size_t blh = (size_t)(b * LL + l);
    float g = wg[blh * HH + h];
    float a0 = lgt[blh * 12 + h * 3 + 0] * itemp;
    float a1 = lgt[blh * 12 + h * 3 + 1] * itemp;
    float a2 = lgt[blh * 12 + h * 3 + 2] * itemp;
    float mx = fmaxf(a0, fmaxf(a1, a2));
    float e0 = expf(a0 - mx), e1 = expf(a1 - mx), e2 = expf(a2 - mx);
    float inv = 1.f / (e0 + e1 + e2);
    float del = dout[(blh * HH + h) * DVC + dv];
    float lm = (e0 * vcur + e1 * lsv + e2 * llv) * inv;
    float val = g * del + (1.f - g) * lm;
    float sq = val * val;
#pragma unroll
    for (int off = 16; off > 0; off >>= 1) sq += __shfl_xor(sq, off, 32);
    if ((threadIdx.x & 31) == 0) red[threadIdx.x >> 5] = sq;
    __syncthreads();
    float tot = 0.f;
#pragma unroll
    for (int wv = 0; wv < 8; wv++) tot += red[wv];
    float rr = rsqrtf(tot * (1.f / 256.f) + 1e-5f);
    __syncthreads();
    oh[(size_t)(b * LL + l) * DD + h * DVC + dv] = (_Float16)(val * rr * rw);
  }
}

// ---------------- host launcher ----------------
extern "C" void kernel_launch(void* const* d_in, const int* in_sizes, int n_in,
                              void* d_out, int out_size, void* d_ws, size_t ws_size,
                              hipStream_t stream) {
  (void)in_sizes; (void)n_in; (void)out_size; (void)ws_size;
  const float* x      = (const float*)d_in[0];
  const float* Wq     = (const float*)d_in[1];
  const float* Wk     = (const float*)d_in[2];
  const float* Wv     = (const float*)d_in[3];
  const float* conv_q = (const float*)d_in[4];
  const float* conv_k = (const float*)d_in[5];
  const float* conv_v = (const float*)d_in[6];
  const float* Wb     = (const float*)d_in[7];
  const float* fir_s  = (const float*)d_in[8];
  const float* fir_l  = (const float*)d_in[9];
  const float* Wg     = (const float*)d_in[10];
  const float* bg     = (const float*)d_in[11];
  const float* Wl     = (const float*)d_in[12];
  const float* bl     = (const float*)d_in[13];
  const float* ltmp   = (const float*)d_in[14];
  const float* rmsw   = (const float*)d_in[15];
  const float* Wo     = (const float*)d_in[16];

  size_t off = 0;
  auto alloc = [&](size_t bytes) -> char* {
    char* p = (char*)d_ws + off;
    off += (bytes + 255) & ~(size_t)255;
    return p;
  };
  const size_t NTOK = (size_t)BB * LL;          // 8192
  _Float16* x_h  = (_Float16*)alloc(NTOK * DD * 2);
  _Float16* wq_h = (_Float16*)alloc((size_t)DD * DD * 2);
  _Float16* wk_h = (_Float16*)alloc((size_t)DD * DD * 2);
  _Float16* wv_h = (_Float16*)alloc((size_t)DD * DD * 2);
  _Float16* wo_h = (_Float16*)alloc((size_t)DD * DD * 2);
  float* q_lin = (float*)alloc(NTOK * DD * 4);
  float* k_lin = (float*)alloc(NTOK * DD * 4);
  float* v_lin = (float*)alloc(NTOK * DD * 4);
  _Float16* q_a = (_Float16*)alloc(NTOK * DD * 2);
  _Float16* k_a = (_Float16*)alloc(NTOK * DD * 2);
  _Float16* v_a = (_Float16*)alloc(NTOK * DD * 2);
  float* betab = (float*)alloc(NTOK * HH * 4);
  float* wgb   = (float*)alloc(NTOK * HH * 4);
  float* lgb   = (float*)alloc(NTOK * 12 * 4);
  float* doutb = (float*)alloc(NTOK * DD * 4);
  _Float16* o_h = (_Float16*)alloc(NTOK * DD * 2);

  // 1) casts to f16
  const size_t nx = NTOK * DD;              // 8.39M
  cast_f16<<<(unsigned)((nx + 255) / 256), 256, 0, stream>>>(x, x_h, nx);
  const size_t nw = (size_t)DD * DD;
  cast_f16<<<(unsigned)((nw + 255) / 256), 256, 0, stream>>>(Wq, wq_h, nw);
  cast_f16<<<(unsigned)((nw + 255) / 256), 256, 0, stream>>>(Wk, wk_h, nw);
  cast_f16<<<(unsigned)((nw + 255) / 256), 256, 0, stream>>>(Wv, wv_h, nw);
  cast_f16<<<(unsigned)((nw + 255) / 256), 256, 0, stream>>>(Wo, wo_h, nw);

  // 2) q/k/v projections (WMMA GEMM, async double-buffered staging)
  dim3 gg(DD / 128, (unsigned)(NTOK / 128));  // (8, 64)
  gemm_f16<<<gg, 256, 0, stream>>>(x_h, wq_h, q_lin, (int)NTOK, DD, DD);
  gemm_f16<<<gg, 256, 0, stream>>>(x_h, wk_h, k_lin, (int)NTOK, DD, DD);
  gemm_f16<<<gg, 256, 0, stream>>>(x_h, wv_h, v_lin, (int)NTOK, DD, DD);

  // 3) causal conv + SiLU -> f16 activations
  unsigned cb = (unsigned)((nx + 255) / 256);
  conv_silu<<<cb, 256, 0, stream>>>(q_lin, conv_q, q_a);
  conv_silu<<<cb, 256, 0, stream>>>(k_lin, conv_k, k_a);
  conv_silu<<<cb, 256, 0, stream>>>(v_lin, conv_v, v_a);

  // 4) beta / global gate / mix logits
  small_proj<<<(BB * LL * 20 + 255) / 256, 256, 0, stream>>>(x, Wb, Wg, bg, Wl, bl,
                                                             betab, wgb, lgb);

  // 5) chunkwise delta rule (WMMA, LDS-resident state, dv-sliced x4)
  (void)hipFuncSetAttribute(reinterpret_cast<const void*>(delta_kernel),
                            hipFuncAttributeMaxDynamicSharedMemorySize, 118784);
  delta_kernel<<<dim3(NSL, HH, BB), 256, 118784, stream>>>(q_a, k_a, v_a, betab, doutb);

  // 6) FIR + gating + RMSNorm -> f16 rows for output projection
  (void)hipFuncSetAttribute(reinterpret_cast<const void*>(fir_mix),
                            hipFuncAttributeMaxDynamicSharedMemorySize, 96288);
  fir_mix<<<dim3(LL / 64, HH, BB), 256, 96288, stream>>>(v_a, doutb, wgb, lgb, fir_s, fir_l,
                                                         ltmp, rmsw, o_h);

  // 7) output projection -> d_out (f32)
  gemm_f16<<<gg, 256, 0, stream>>>(o_h, wo_h, (float*)d_out, (int)NTOK, DD, DD);
}
